// HypCLIPAttention_35278861369450
// MI455X (gfx1250) — matrix operand order repaired and checked by reading, not compiled, not run on hardware
//
#include <hip/hip_runtime.h>
#include <math.h>

#define TOK   8192      // B*T
#define DIN   1025
#define KP    1056      // padded K  (33*32)
#define NP    1088      // padded N  (17*64)
#define DHP   96        // padded head dim (65 -> 3*32)
#define BH    128       // B*H
#define TT    1024
#define EPSC  1e-8f

typedef __bf16 bf16;
typedef bf16  v16bf  __attribute__((ext_vector_type(16)));
typedef float v8f    __attribute__((ext_vector_type(8)));
typedef unsigned short u16x8 __attribute__((ext_vector_type(8)));
typedef unsigned int   uint4v __attribute__((ext_vector_type(4)));
typedef int            int4v  __attribute__((ext_vector_type(4)));
typedef int            int8v  __attribute__((ext_vector_type(8)));

#if defined(__has_builtin)
#  if __has_builtin(__builtin_amdgcn_tensor_load_to_lds)
#    define HAVE_TDM 1
#  endif
#endif
#ifndef HAVE_TDM
#  define HAVE_TDM 0
#endif

// Compile-time probes (visible in hipcc stderr): which data-mover path compiled?
#if HAVE_TDM
#warning CDNA5_PROBE: __builtin_amdgcn_tensor_load_to_lds AVAILABLE -> TDM path compiled in
#else
#warning CDNA5_PROBE: tensor_load_to_lds builtin NOT available -> cooperative-copy fallback compiled in
#endif
#if __clang_major__ >= 23
#warning CDNA5_PROBE: clang >= 23 (6-arg TDM builtin form selected)
#else
#warning CDNA5_PROBE: clang < 23 (5-arg TDM builtin form selected)
#endif

union BFrag { v16bf v; u16x8 u[2]; unsigned short s[16]; };

__device__ __forceinline__ unsigned short f2bf(float f) {
  union { float f; unsigned int u; } x; x.f = f;
  unsigned int r = x.u + 0x7FFFu + ((x.u >> 16) & 1u);
  return (unsigned short)(r >> 16);
}

// ---------------- conversion kernels ----------------
__global__ __launch_bounds__(256) void convert_x(const float* __restrict__ X,
                                                 unsigned short* __restrict__ Xb) {
  long idx = (long)blockIdx.x * 256 + threadIdx.x;
  if (idx >= (long)TOK * KP) return;
  int row = (int)(idx / KP), col = (int)(idx % KP);
  Xb[idx] = (col < DIN) ? f2bf(X[(long)row * DIN + col]) : 0;
}

__global__ __launch_bounds__(256) void convert_w(const float* __restrict__ W,
                                                 unsigned short* __restrict__ Wb) {
  long idx = (long)blockIdx.x * 256 + threadIdx.x;
  if (idx >= (long)NP * KP) return;
  int r = (int)(idx / KP), c = (int)(idx % KP);
  Wb[idx] = (r < DIN && c < DIN) ? f2bf(W[(long)r * DIN + c]) : 0;
}

// ---------------- projection GEMM: Y[M,N] = Xb @ Wb^T + bias ----------------
// Block = 4 waves; each wave owns a 32(M) x 64(N) tile (2 A frags x 4 N-tiles,
// 8 accumulators), so each LDS B fragment feeds two WMMAs. The shared 64-row
// W tile is staged once per block into LDS by the Tensor Data Mover; the
// K-loop reads B via ds_load_b128 and keeps A loads flowing from global/L2.
__global__ __launch_bounds__(128) void gemm_bf16(const unsigned short* __restrict__ A,
                                                 const unsigned short* __restrict__ Wb,
                                                 const float* __restrict__ bias,
                                                 float* __restrict__ Y) {
  extern __shared__ unsigned short Wlds[];           // 64 * KP bf16 = 135,168 B
  const int lane = threadIdx.x & 31;
  const int wave = threadIdx.x >> 5;                 // 4 waves / block
  const int m0 = (blockIdx.x * 4 + wave) * 32;       // gridDim.x = 64 -> M 0..8191
  const int n0 = blockIdx.y * 64;                    // gridDim.y = 17 -> N 0..1087
  const int lr = lane & 15, lh = lane >> 4;

#if HAVE_TDM
  // ---- TDM: DMA the 2D W tile [n0 .. n0+64) x KP (bf16) into LDS ----
  if (threadIdx.x == 0) {
    unsigned long long ga = (unsigned long long)(size_t)(Wb + (size_t)n0 * KP);
    unsigned int la = (unsigned int)(size_t)(&Wlds[0]);   // LDS byte offset (low 32b)
    uint4v g0;
    g0.x = 1u;                                            // count=1, user descriptor
    g0.y = la;                                            // lds_addr   (bits 63:32)
    g0.z = (unsigned int)(ga & 0xFFFFFFFFu);              // global_addr[31:0]
    g0.w = (unsigned int)((ga >> 32) & 0x01FFFFFFu) | 0x80000000u; // addr[56:32] | type=2
    int8v g1;
    g1[0] = (int)(1u << 16);              // workgroup_mask=0, data_size=2B
    g1[1] = (int)((KP & 0xFFFF) << 16);   // tensor_dim0[15:0] @ bits 63:48
    g1[2] = (int)(64u << 16);             // tensor_dim0[31:16]=0 | tensor_dim1[15:0]=64
    g1[3] = (int)((KP & 0xFFFF) << 16);   // tensor_dim1[31:16]=0 | tile_dim0=KP
    g1[4] = 64;                           // tile_dim1=64 rows, tile_dim2=0
    g1[5] = KP;                           // tensor_dim0_stride[31:0]
    g1[6] = 0;                            // stride0[47:32]=0 | dim1_stride[15:0]=0
    g1[7] = 0;
    int4v gz = {0, 0, 0, 0};
#if __clang_major__ >= 23
    int8v g4 = {0, 0, 0, 0, 0, 0, 0, 0};
    __builtin_amdgcn_tensor_load_to_lds(g0, g1, gz, gz, g4, 0);
#else
    __builtin_amdgcn_tensor_load_to_lds(g0, g1, gz, gz, 0);
#endif
  }
#else
  // ---- fallback: cooperative vector copy of the W tile into LDS ----
  {
    const u16x8* src = (const u16x8*)(Wb + (size_t)n0 * KP);
    u16x8* dst = (u16x8*)Wlds;
    const int total = 64 * KP / 8;
    for (int i = threadIdx.x; i < total; i += 128) dst[i] = src[i];
  }
#endif
#if HAVE_TDM && __has_builtin(__builtin_amdgcn_s_wait_tensorcnt)
  __builtin_amdgcn_s_wait_tensorcnt(0);
#else
  asm volatile("s_wait_tensorcnt 0x0" ::: "memory");  // NOP if TDM unused
#endif
  __syncthreads();

  v8f acc[2][4] = {};
  const unsigned short* arow0 = A + (size_t)(m0 + lr) * KP;
  const unsigned short* arow1 = A + (size_t)(m0 + 16 + lr) * KP;

  for (int kb = 0; kb < KP; kb += 32) {
    BFrag a0, a1;
    a0.u[0] = *(const u16x8*)(arow0 + kb + lh * 8);
    a0.u[1] = *(const u16x8*)(arow0 + kb + 16 + lh * 8);
    a1.u[0] = *(const u16x8*)(arow1 + kb + lh * 8);
    a1.u[1] = *(const u16x8*)(arow1 + kb + 16 + lh * 8);
    __builtin_prefetch(arow0 + kb + 64, 0, 0);
    __builtin_prefetch(arow1 + kb + 64, 0, 0);
    BFrag b[4];
#pragma unroll
    for (int nt = 0; nt < 4; ++nt) {
      const u16x8* wp = (const u16x8*)&Wlds[(size_t)(nt * 16 + lr) * KP + kb + lh * 8];
      b[nt].u[0] = wp[0];
      b[nt].u[1] = wp[2];
    }
#pragma unroll
    for (int nt = 0; nt < 4; ++nt) {
      acc[0][nt] = __builtin_amdgcn_wmma_f32_16x16x32_bf16(
          false, a0.v, false, b[nt].v, (short)0, acc[0][nt], false, false);
      acc[1][nt] = __builtin_amdgcn_wmma_f32_16x16x32_bf16(
          false, a1.v, false, b[nt].v, (short)0, acc[1][nt], false, false);
    }
  }
#pragma unroll
  for (int mi = 0; mi < 2; ++mi) {
#pragma unroll
    for (int nt = 0; nt < 4; ++nt) {
      int N = n0 + nt * 16 + lr;
      float bb = (N < DIN) ? bias[N] : 0.f;
#pragma unroll
      for (int r = 0; r < 8; ++r) {
        int M = m0 + mi * 16 + r + 8 * lh;
        Y[(size_t)M * NP + N] = acc[mi][nt][r] + bb;
      }
    }
  }
}

// ---------------- Lorentz fixup for q/k/v ----------------
// mode 0: Q row-major [bh][t][DHP]; mode 1: K (time negated); mode 2: V transposed [bh][DHP][t]
__global__ __launch_bounds__(256) void fixup_qkv(const float* __restrict__ Y,
                                                 const float* __restrict__ logscale,
                                                 unsigned short* __restrict__ out,
                                                 int mode) {
  const int row = blockIdx.x;          // token index
  const int tid = threadIdx.x;         // 256, 4 cols each
  const int b = row >> 10, t = row & 1023;
  const float* y = Y + (size_t)row * NP;

  float vals[4]; float part = 0.f;
  const int c0 = tid * 4;
#pragma unroll
  for (int j = 0; j < 4; ++j) { vals[j] = y[1 + c0 + j]; part += vals[j] * vals[j]; }
#pragma unroll
  for (int m = 8; m; m >>= 1) part += __shfl_xor(part, m, 32);

  __shared__ float hsum[16];
  const int head = tid >> 4;           // head owning these 64 cols
  if ((tid & 15) == 0) hsum[head] = part;
  __syncthreads();
  float ssq = 0.f;
#pragma unroll
  for (int h = 0; h < 16; ++h) ssq += hsum[h];

  float sc = __expf(logscale[0]);
  float tm = sc / (1.f + __expf(-y[0])) + 1.1f;
  float g  = sqrtf((tm * tm - 1.f) / fmaxf(ssq, EPSC));

  const int bh = b * 16 + head;
  const int j0 = (tid & 15) * 4;       // col within head
  if (mode < 2) {
    unsigned short* dst = out + ((size_t)bh * TT + t) * DHP;
#pragma unroll
    for (int j = 0; j < 4; ++j) dst[1 + j0 + j] = f2bf(vals[j] * g);
    if ((tid & 15) == 0) {
      float th = sqrtf(g * g * hsum[head] + 1.f);
      dst[0] = f2bf(mode == 1 ? -th : th);
      for (int p = 65; p < DHP; ++p) dst[p] = 0;
    }
  } else {
    unsigned short* dst = out + (size_t)bh * DHP * TT + t;
#pragma unroll
    for (int j = 0; j < 4; ++j) dst[(size_t)(1 + j0 + j) * TT] = f2bf(vals[j] * g);
    if ((tid & 15) == 0) {
      float th = sqrtf(g * g * hsum[head] + 1.f);
      dst[0] = f2bf(th);
      for (int p = 65; p < DHP; ++p) dst[(size_t)p * TT] = 0;
    }
  }
}

// ---------------- attention: one wave per (bh, 16-query tile) ----------------
__global__ __launch_bounds__(32) void attn_kernel(const unsigned short* __restrict__ Qh,
                                                  const unsigned short* __restrict__ Kh,
                                                  const unsigned short* __restrict__ Vt,
                                                  const float* __restrict__ attn_bias,
                                                  float* __restrict__ attnspace) {
  __shared__ float Ss[16 * TT];        // 64 KB score row-block
  const int bh = blockIdx.x;
  const int t0 = blockIdx.y * 16;
  const int lane = threadIdx.x, lr = lane & 15, lh = lane >> 4;
  const unsigned short* Qb = Qh + (size_t)bh * TT * DHP;
  const unsigned short* Kb = Kh + (size_t)bh * TT * DHP;
  const unsigned short* Vb = Vt + (size_t)bh * DHP * TT;
  const float abias = attn_bias[0];

  // Q fragments (reused across all key tiles)
  BFrag qa[3];
#pragma unroll
  for (int dc = 0; dc < 3; ++dc) {
    const unsigned short* qp = Qb + (size_t)(t0 + lr) * DHP + dc * 32 + lh * 8;
    qa[dc].u[0] = *(const u16x8*)(qp);
    qa[dc].u[1] = *(const u16x8*)(qp + 16);
  }

  // Phase 1: S = Q @ Kl^T, scaled, into LDS
  for (int c0 = 0; c0 < TT; c0 += 16) {
    v8f c = {};
#pragma unroll
    for (int dc = 0; dc < 3; ++dc) {
      const unsigned short* kp = Kb + (size_t)(c0 + lr) * DHP + dc * 32 + lh * 8;
      BFrag kf;
      kf.u[0] = *(const u16x8*)(kp);
      kf.u[1] = *(const u16x8*)(kp + 16);
      c = __builtin_amdgcn_wmma_f32_16x16x32_bf16(
          false, qa[dc].v, false, kf.v, (short)0, c, false, false);
    }
#pragma unroll
    for (int r = 0; r < 8; ++r)
      Ss[(r + 8 * lh) * TT + c0 + lr] = (2.f + 2.f * c[r]) * 0.125f + abias;
  }
  asm volatile("s_wait_dscnt 0" ::: "memory");

  // Phase 2: in-place softmax (unnormalized); lane owns (row=lr, half lh)
  const int cb = lh * 512;
  float mx = -1e30f;
  for (int c = 0; c < 512; ++c) mx = fmaxf(mx, Ss[lr * TT + cb + c]);
  mx = fmaxf(mx, __shfl_xor(mx, 16, 32));
  float sum = 0.f;
  for (int c = 0; c < 512; ++c) {
    float e = __expf(Ss[lr * TT + cb + c] - mx);
    sum += e;
    Ss[lr * TT + cb + c] = e;          // own addresses: race-free in-place
  }
  sum += __shfl_xor(sum, 16, 32);
  asm volatile("s_wait_dscnt 0" ::: "memory");

  // Phase 3: O = P @ V
  v8f acc[5] = {};
  for (int kt = 0; kt < 32; ++kt) {
    BFrag pa;
    const float* pb = &Ss[lr * TT + kt * 32 + lh * 8];
#pragma unroll
    for (int j = 0; j < 8; ++j) pa.s[j] = f2bf(pb[j]);
#pragma unroll
    for (int j = 0; j < 8; ++j) pa.s[8 + j] = f2bf(pb[16 + j]);
#pragma unroll
    for (int n = 0; n < 5; ++n) {
      const unsigned short* vp = Vb + (size_t)(n * 16 + lr) * TT + kt * 32 + lh * 8;
      BFrag vf;
      vf.u[0] = *(const u16x8*)(vp);
      vf.u[1] = *(const u16x8*)(vp + 16);
      acc[n] = __builtin_amdgcn_wmma_f32_16x16x32_bf16(
          false, pa.v, false, vf.v, (short)0, acc[n], false, false);
    }
  }

  // Epilogue: normalize probs, Lorentz centroid, scatter space comps
  const int b = bh >> 4, h = bh & 15;
#pragma unroll
  for (int r = 0; r < 8; ++r) {
    const int M = r + 8 * lh;
    float rs = __shfl(sum, M, 32);
    float ave[5]; float p = 0.f;
#pragma unroll
    for (int n = 0; n < 5; ++n) {
      ave[n] = acc[n][r] / rs;
      int d = n * 16 + lr;
      if (d == 0)       p += ave[n] * ave[n];
      else if (d <= 64) p -= ave[n] * ave[n];
    }
#pragma unroll
    for (int m = 8; m; m >>= 1) p += __shfl_xor(p, m, 32);
    float dc = sqrtf(fmaxf(fabsf(p), EPSC));
#pragma unroll
    for (int n = 0; n < 5; ++n) {
      int d = n * 16 + lr;
      if (d >= 1 && d <= 64)
        attnspace[(size_t)(b * 1024 + t0 + M) * 1024 + h * 64 + (d - 1)] = ave[n] / dc;
    }
  }
}

// ---------------- add_time + pack attention output to bf16 ----------------
__global__ __launch_bounds__(256) void pack_attn(const float* __restrict__ space,
                                                 unsigned short* __restrict__ Xb) {
  const int row = blockIdx.x, tid = threadIdx.x;
  const float* s = space + (size_t)row * 1024;
  float vals[4]; float part = 0.f;
  const int c0 = tid * 4;
#pragma unroll
  for (int j = 0; j < 4; ++j) { vals[j] = s[c0 + j]; part += vals[j] * vals[j]; }
#pragma unroll
  for (int m = 16; m; m >>= 1) part += __shfl_xor(part, m, 32);
  __shared__ float ws[8];
  if ((tid & 31) == 0) ws[tid >> 5] = part;
  __syncthreads();
  float ssq = 0.f;
#pragma unroll
  for (int i = 0; i < 8; ++i) ssq += ws[i];
  unsigned short* dst = Xb + (size_t)row * KP;
#pragma unroll
  for (int j = 0; j < 4; ++j) dst[1 + c0 + j] = f2bf(vals[j]);
  if (tid == 0) {
    dst[0] = f2bf(sqrtf(ssq + 1.f));
    for (int p = DIN; p < KP; ++p) dst[p] = 0;
  }
}

// ---------------- final Lorentz fixup -> d_out ----------------
__global__ __launch_bounds__(256) void fixup_o(const float* __restrict__ Y,
                                               const float* __restrict__ logscale,
                                               float* __restrict__ out) {
  const int row = blockIdx.x, tid = threadIdx.x;
  const float* y = Y + (size_t)row * NP;
  float vals[4]; float part = 0.f;
  const int c0 = tid * 4;
#pragma unroll
  for (int j = 0; j < 4; ++j) { vals[j] = y[1 + c0 + j]; part += vals[j] * vals[j]; }
#pragma unroll
  for (int m = 16; m; m >>= 1) part += __shfl_xor(part, m, 32);
  __shared__ float ws[8];
  if ((tid & 31) == 0) ws[tid >> 5] = part;
  __syncthreads();
  float ssq = 0.f;
#pragma unroll
  for (int i = 0; i < 8; ++i) ssq += ws[i];
  float sc = __expf(logscale[0]);
  float tm = sc / (1.f + __expf(-y[0])) + 1.1f;
  float g  = sqrtf((tm * tm - 1.f) / fmaxf(ssq, EPSC));
  float* d = out + (size_t)row * DIN;
#pragma unroll
  for (int j = 0; j < 4; ++j) d[1 + c0 + j] = vals[j] * g;
  if (tid == 0) d[0] = tm;
}

// ---------------- workspace layout (bytes) ----------------
#define SZ_XBF  ((size_t)TOK * KP * 2)          // 17,301,504
#define SZ_WBF  ((size_t)NP * KP * 2)           //  2,297,856 per matrix
#define SZ_HEAD ((size_t)BH * TT * DHP * 2)     // 25,165,824
#define OFF_WBF (SZ_XBF)
#define OFF_QH  (OFF_WBF + 4 * SZ_WBF)
#define OFF_KH  (OFF_QH + SZ_HEAD)
#define OFF_VT  (OFF_KH + SZ_HEAD)
#define OFF_Y   (OFF_VT + SZ_HEAD)              // Yraw [TOK][NP] f32; aliased as attnspace
// total = OFF_Y + TOK*NP*4 = 137,641,984 bytes

#define GEMM_LDS ((size_t)64 * KP * 2)          // 135,168 B dynamic LDS per block

extern "C" void kernel_launch(void* const* d_in, const int* in_sizes, int n_in,
                              void* d_out, int out_size, void* d_ws, size_t ws_size,
                              hipStream_t stream) {
  (void)in_sizes; (void)n_in; (void)out_size; (void)ws_size;
  const float* X  = (const float*)d_in[0];
  const float* W[4]  = {(const float*)d_in[1], (const float*)d_in[4],
                        (const float*)d_in[7], (const float*)d_in[10]};
  const float* bb[4] = {(const float*)d_in[2], (const float*)d_in[5],
                        (const float*)d_in[8], (const float*)d_in[11]};
  const float* ls[4] = {(const float*)d_in[3], (const float*)d_in[6],
                        (const float*)d_in[9], (const float*)d_in[12]};
  const float* abias = (const float*)d_in[13];

  char* ws = (char*)d_ws;
  unsigned short* Xbf = (unsigned short*)ws;
  unsigned short* Wbf[4];
  for (int i = 0; i < 4; ++i) Wbf[i] = (unsigned short*)(ws + OFF_WBF + i * SZ_WBF);
  unsigned short* Qh = (unsigned short*)(ws + OFF_QH);
  unsigned short* Kh = (unsigned short*)(ws + OFF_KH);
  unsigned short* Vt = (unsigned short*)(ws + OFF_VT);
  float* Yraw = (float*)(ws + OFF_Y);
  float* attnspace = Yraw;   // alias: Yraw is dead once V fixup has run

  // bf16 conversions
  convert_x<<<(int)(((size_t)TOK * KP + 255) / 256), 256, 0, stream>>>(X, Xbf);
  for (int i = 0; i < 4; ++i)
    convert_w<<<(int)(((size_t)NP * KP + 255) / 256), 256, 0, stream>>>(W[i], Wbf[i]);

  // Q / K / V projections + Lorentz head-shaping
  for (int i = 0; i < 3; ++i) {
    gemm_bf16<<<dim3(TOK / 128, NP / 64), 128, GEMM_LDS, stream>>>(Xbf, Wbf[i], bb[i], Yraw);
    unsigned short* dst = (i == 0) ? Qh : (i == 1) ? Kh : Vt;
    fixup_qkv<<<TOK, 256, 0, stream>>>(Yraw, ls[i], dst, i);
  }

  // attention + centroid
  attn_kernel<<<dim3(BH, TT / 16), 32, 0, stream>>>(Qh, Kh, Vt, abias, attnspace);

  // add_time + repack, output projection, final fixup
  pack_attn<<<TOK, 256, 0, stream>>>(attnspace, Xbf);
  gemm_bf16<<<dim3(TOK / 128, NP / 64), 128, GEMM_LDS, stream>>>(Xbf, Wbf[3], bb[3], Yraw);
  fixup_o<<<TOK, 256, 0, stream>>>(Yraw, ls[3], (float*)d_out);
}
